// RGAT_43258910605913
// MI455X (gfx1250) — compile-verified, matrix operand order checked by reading, not run on hardware
//
#include <hip/hip_runtime.h>
#include <cmath>
#include <cstdint>

// ---------- CDNA5 WMMA vector types ----------
typedef __attribute__((ext_vector_type(16))) __bf16 v16bf;
typedef __attribute__((ext_vector_type(8)))  __bf16 v8bf;
typedef __attribute__((ext_vector_type(4)))  __bf16 v4bf;
typedef __attribute__((ext_vector_type(8)))  float  v8f;

#define NEG_SLOPE 0.2f

// Async global->LDS staging (gfx1250), guarded so we fall back cleanly if the
// toolchain doesn't declare the builtins.
#if __has_builtin(__builtin_amdgcn_global_load_async_to_lds_b128) && \
    __has_builtin(__builtin_amdgcn_s_wait_asynccnt)
#define USE_ASYNC_LDS 1
// builtin signature (per clang-22 diagnostic): (v4i addrspace(1)*, v4i addrspace(3)*, imm, imm)
typedef int v4i_async __attribute__((__vector_size__(4 * sizeof(int))));
typedef __attribute__((address_space(1))) v4i_async* g_v4i_p;
typedef __attribute__((address_space(3))) v4i_async* l_v4i_p;
#else
#define USE_ASYNC_LDS 0
#endif

// ---------------- utility kernels ----------------
__global__ void cvt_f32_bf16(const float* __restrict__ src, __bf16* __restrict__ dst, size_t n) {
  size_t i = (size_t)blockIdx.x * blockDim.x + threadIdx.x;
  if (i < n) dst[i] = (__bf16)src[i];
}

// W[r][k][n] (fp32) -> wbT[r][n][k] (bf16): coalesced read, tiny tensor
__global__ void cvt_transpose_w(const float* __restrict__ w, __bf16* __restrict__ wbT, int R) {
  size_t i = (size_t)blockIdx.x * blockDim.x + threadIdx.x;
  if (i >= (size_t)R * 16384) return;
  int r = (int)(i >> 14);
  int k = (int)((i >> 7) & 127);
  int n = (int)(i & 127);
  wbT[((size_t)r << 14) + ((size_t)n << 7) + k] = (__bf16)w[i];
}

__global__ void fill_f32(float* __restrict__ p, float v, size_t n) {
  size_t i = (size_t)blockIdx.x * blockDim.x + threadIdx.x;
  if (i < n) p[i] = v;
}

// ---------------- per-relation transform GEMM (WMMA bf16) ----------------
// xr[r] = Xb @ W[r]   (N x 128) @ (128 x 128), bf16 in, fp32 accum, bf16 out
// epilogue: qv[r,n] = xr[r,n,:] . q   kv[r,n] = xr[r,n,:] . k
// B fragments (W columns) are loop-invariant: loaded once from pre-transposed
// global W into VGPRs (2 x global_load_b128 per fragment).
__global__ __launch_bounds__(256) void rgat_gemm_wmma(
    const __bf16* __restrict__ xb, const __bf16* __restrict__ wbT,
    const float* __restrict__ q, const float* __restrict__ k,
    __bf16* __restrict__ xr, float* __restrict__ qv, float* __restrict__ kv,
    int N) {
  constexpr int LP = 136;              // padded LDS row stride (bank-conflict-free)
  __shared__ __bf16 Xlds[16 * LP];     // 16 rows x 128 K (padded)
  __shared__ __bf16 Stg[16 * LP];      // staged output tile (padded)
  __shared__ float  qlds[128];
  __shared__ float  klds[128];

  const int tid  = threadIdx.x;
  const int lane = tid & 31;
  const int wave = tid >> 5;           // 8 waves -> 8 column tiles
  const int r    = blockIdx.y;
  const int M    = lane & 15;
  const int kb   = (lane >> 4) ? 8 : 0;  // K-half per documented 16-bit layout
  const int ncol = wave * 16 + M;

  __bf16* xrr = xr + (size_t)r * N * 128;
  float*  qvr = qv + (size_t)r * N;
  float*  kvr = kv + (size_t)r * N;

  // ---- load loop-invariant B fragments into registers ----
  const __bf16* wcol = wbT + (((size_t)r << 7) + ncol) * 128;
  v16bf bfrag[4];
#pragma unroll
  for (int kk = 0; kk < 4; ++kk) {
    v8bf lo = *reinterpret_cast<const v8bf*>(wcol + kk * 32 + kb);
    v8bf hi = *reinterpret_cast<const v8bf*>(wcol + kk * 32 + kb + 16);
    bfrag[kk] = __builtin_shufflevector(lo, hi, 0, 1, 2, 3, 4, 5, 6, 7,
                                        8, 9, 10, 11, 12, 13, 14, 15);
  }
  if (tid < 128) { qlds[tid] = q[tid]; klds[tid] = k[tid]; }

  const int nRowBlocks = N >> 4;
  const int row16 = tid >> 4;          // 0..15: row within tile
  const int grp   = tid & 15;          // 0..15: 8-elem group within row

  for (int rb = blockIdx.x; rb < nRowBlocks; rb += gridDim.x) {
    const int rowBase = rb << 4;
    // stage 16x128 X tile into padded LDS
#if USE_ASYNC_LDS
    __builtin_amdgcn_global_load_async_to_lds_b128(
        (g_v4i_p)(xb + (size_t)(rowBase + row16) * 128 + grp * 8),
        (l_v4i_p)&Xlds[row16 * LP + grp * 8],
        0, 0);
    __builtin_amdgcn_s_wait_asynccnt(0);
#else
    *reinterpret_cast<v8bf*>(&Xlds[row16 * LP + grp * 8]) =
        *reinterpret_cast<const v8bf*>(xb + (size_t)(rowBase + row16) * 128 + grp * 8);
#endif
    __syncthreads();

    v8f c = {};
#pragma unroll
    for (int kk = 0; kk < 4; ++kk) {
      v8bf alo = *reinterpret_cast<const v8bf*>(&Xlds[M * LP + kk * 32 + kb]);
      v8bf ahi = *reinterpret_cast<const v8bf*>(&Xlds[M * LP + kk * 32 + kb + 16]);
      v16bf a = __builtin_shufflevector(alo, ahi, 0, 1, 2, 3, 4, 5, 6, 7,
                                        8, 9, 10, 11, 12, 13, 14, 15);
      c = __builtin_amdgcn_wmma_f32_16x16x32_bf16(false, a, false, bfrag[kk],
                                                  (short)0, c, false, false);
    }

    // stage D tile: VGPR i holds row i (lanes 0-15) / i+8 (lanes 16-31), col = ncol
    const int rowoff = (lane >> 4) ? 8 : 0;
#pragma unroll
    for (int i = 0; i < 8; ++i)
      Stg[(i + rowoff) * LP + ncol] = (__bf16)c[i];
    __syncthreads();

    // coalesced b128 store of bf16 tile to xr
    *reinterpret_cast<v8bf*>(xrr + (size_t)(rowBase + row16) * 128 + grp * 8) =
        *reinterpret_cast<const v8bf*>(&Stg[row16 * LP + grp * 8]);

    // qv/kv epilogue: one row per thread (16 rows)
    if (tid < 16) {
      float sq = 0.f, sk = 0.f;
#pragma unroll 8
      for (int cc = 0; cc < 128; ++cc) {
        float v = (float)Stg[tid * LP + cc];
        sq += v * qlds[cc];
        sk += v * klds[cc];
      }
      qvr[rowBase + tid] = sq;
      kvr[rowBase + tid] = sk;
    }
    __syncthreads();   // protect Xlds/Stg reuse
  }
}

// ---------------- edge kernels ----------------
__global__ void edge_alpha_max(const int* __restrict__ src, const int* __restrict__ dst,
                               const int* __restrict__ et,
                               const float* __restrict__ qv, const float* __restrict__ kv,
                               float* __restrict__ alpha, float* __restrict__ amax,
                               int E, int N) {
  int e = blockIdx.x * blockDim.x + threadIdx.x;
  if (e >= E) return;
  int s = src[e], d = dst[e], t = et[e];
  float a = qv[(size_t)t * N + d] + kv[(size_t)t * N + s];
  a = (a >= 0.f) ? a : NEG_SLOPE * a;
  alpha[e] = a;
  __hip_atomic_fetch_max(&amax[d], a, __ATOMIC_RELAXED, __HIP_MEMORY_SCOPE_AGENT);
}

// one wave per edge. denominator is factored out of the edge sum:
//   out[d] = (sum_e ex_e * xr_row_e) / denom[d]
// so this single pass computes ex, accumulates denom (lane 0) and scatters
// the un-normalized weighted message. Division happens in bias_relu.
__global__ __launch_bounds__(256) void edge_scatter_exp(
    const int* __restrict__ src, const int* __restrict__ dst, const int* __restrict__ et,
    const float* __restrict__ alpha, const float* __restrict__ amax,
    const __bf16* __restrict__ xr, float* __restrict__ denom, float* __restrict__ acc,
    int E, int N) {
  int gw   = blockIdx.x * 8 + (threadIdx.x >> 5);
  int lane = threadIdx.x & 31;
  if (gw >= E) return;
  int e = __builtin_amdgcn_readfirstlane(gw);   // wave-uniform -> SGPR
  int s = src[e], d = dst[e], t = et[e];
  const __bf16* row = xr + ((size_t)t * N + s) * 128;
  __builtin_prefetch(row, 0, 1);                // global_prefetch_b8
  float m  = amax[d];
  float mu = (m < -3.0e38f) ? 0.f : m;          // isfinite(amax) ? amax : 0
  float ex = __expf(alpha[e] - mu);
  if (lane == 0) atomicAdd(&denom[d], ex);
  v4bf v = reinterpret_cast<const v4bf*>(row)[lane];
  float* o = acc + (size_t)d * 128 + lane * 4;
#pragma unroll
  for (int j = 0; j < 4; ++j) atomicAdd(o + j, (float)v[j] * ex);
}

__global__ void bias_relu(const float* __restrict__ acc, const float* __restrict__ denom,
                          const float* __restrict__ b,
                          float* __restrict__ h, __bf16* __restrict__ hb, size_t n) {
  size_t i = (size_t)blockIdx.x * blockDim.x + threadIdx.x;
  if (i >= n) return;
  float v = acc[i] / (denom[i >> 7] + 1e-16f) + b[i & 127];
  v = v > 0.f ? v : 0.f;
  h[i] = v;
  hb[i] = (__bf16)v;
}

// ---------------- pooling + head ----------------
__global__ __launch_bounds__(256) void pool_kernel(const float* __restrict__ h,
                                                   float* __restrict__ psum,
                                                   float* __restrict__ pmax, int N) {
  __shared__ float ssum[256];
  __shared__ float smax[256];
  int col  = threadIdx.x & 127;
  int half = threadIdx.x >> 7;
  float ls = 0.f, lm = -3.4e38f;
  for (int n = blockIdx.x * 2 + half; n < N; n += gridDim.x * 2) {
    float v = h[(size_t)n * 128 + col];
    ls += v;
    lm = fmaxf(lm, v);
  }
  ssum[threadIdx.x] = ls;
  smax[threadIdx.x] = lm;
  __syncthreads();
  if (threadIdx.x < 128) {
    ls = ssum[threadIdx.x] + ssum[threadIdx.x + 128];
    lm = fmaxf(smax[threadIdx.x], smax[threadIdx.x + 128]);
    atomicAdd(&psum[col], ls);
    __hip_atomic_fetch_max(&pmax[col], lm, __ATOMIC_RELAXED, __HIP_MEMORY_SCOPE_AGENT);
  }
}

__global__ __launch_bounds__(256) void head_kernel(
    const float* __restrict__ psum, const float* __restrict__ pmax,
    const float* __restrict__ fc1w, const float* __restrict__ fc1b,
    const float* __restrict__ fc2w, const float* __restrict__ fc2b,
    float* __restrict__ out, int N) {
  __shared__ float g[256];
  __shared__ float gg[128];
  __shared__ float red[128];
  int t = threadIdx.x;
  if (t < 128) g[t] = tanhf(psum[t] / (float)N);
  else         g[t] = tanhf(pmax[t - 128]);
  __syncthreads();
  if (t < 128) {
    float a = fc1b[t];
    for (int i = 0; i < 256; ++i) a += g[i] * fc1w[i * 128 + t];
    gg[t] = a > 0.f ? a : 0.f;
  }
  __syncthreads();
  if (t < 128) red[t] = gg[t] * fc2w[t];
  __syncthreads();
  for (int s = 64; s > 0; s >>= 1) {
    if (t < s) red[t] += red[t + s];
    __syncthreads();
  }
  if (t == 0) {
    float z = red[0] + fc2b[0];
    out[0] = 1.f / (1.f + __expf(-z));
  }
}

// ---------------- host launcher ----------------
extern "C" void kernel_launch(void* const* d_in, const int* in_sizes, int n_in,
                              void* d_out, int out_size, void* d_ws, size_t ws_size,
                              hipStream_t stream) {
  (void)n_in; (void)out_size; (void)ws_size;
  const float* x    = (const float*)d_in[0];
  const int*   ei   = (const int*)d_in[1];
  const int*   et   = (const int*)d_in[2];
  const float* w0   = (const float*)d_in[3];
  const float* q0   = (const float*)d_in[4];
  const float* k0   = (const float*)d_in[5];
  const float* b0   = (const float*)d_in[6];
  const float* w1   = (const float*)d_in[7];
  const float* q1   = (const float*)d_in[8];
  const float* k1   = (const float*)d_in[9];
  const float* b1   = (const float*)d_in[10];
  const float* fc1w = (const float*)d_in[11];
  const float* fc1b = (const float*)d_in[12];
  const float* fc2w = (const float*)d_in[13];
  const float* fc2b = (const float*)d_in[14];

  const int N = in_sizes[0] / 128;
  const int E = in_sizes[2];
  const int R = in_sizes[3] / (128 * 128);
  const int* src = ei;
  const int* dst = ei + E;

  // carve workspace
  char*  ws  = (char*)d_ws;
  size_t off = 0;
  auto alloc = [&](size_t bytes) -> char* {
    char* p = ws + off;
    off += (bytes + 255) & ~(size_t)255;
    return p;
  };
  __bf16* xb    = (__bf16*)alloc((size_t)N * 128 * 2);
  __bf16* hb    = (__bf16*)alloc((size_t)N * 128 * 2);
  __bf16* wbT   = (__bf16*)alloc((size_t)R * 128 * 128 * 2);
  __bf16* xr    = (__bf16*)alloc((size_t)R * N * 128 * 2);
  float*  qv    = (float*)alloc((size_t)R * N * 4);
  float*  kv    = (float*)alloc((size_t)R * N * 4);
  float*  alp   = (float*)alloc((size_t)E * 4);
  float*  amax  = (float*)alloc((size_t)N * 4);
  float*  denom = (float*)alloc((size_t)N * 4);
  float*  acc   = (float*)alloc((size_t)N * 128 * 4);
  float*  h     = (float*)alloc((size_t)N * 128 * 4);
  float*  psum  = (float*)alloc(128 * 4);
  float*  pmax  = (float*)alloc(128 * 4);

  auto cdiv = [](size_t a, size_t b) -> unsigned int {
    return (unsigned int)((a + b - 1) / b);
  };

  auto run_layer = [&](const __bf16* xin, const float* w, const float* q,
                       const float* k, const float* b) {
    cvt_transpose_w<<<cdiv((size_t)R * 16384, 256), 256, 0, stream>>>(w, wbT, R);
    rgat_gemm_wmma<<<dim3(512, R), 256, 0, stream>>>(xin, wbT, q, k, xr, qv, kv, N);
    fill_f32<<<cdiv(N, 256), 256, 0, stream>>>(amax, -INFINITY, (size_t)N);
    fill_f32<<<cdiv(N, 256), 256, 0, stream>>>(denom, 0.f, (size_t)N);
    fill_f32<<<cdiv((size_t)N * 128, 256), 256, 0, stream>>>(acc, 0.f, (size_t)N * 128);
    edge_alpha_max<<<cdiv(E, 256), 256, 0, stream>>>(src, dst, et, qv, kv, alp, amax, E, N);
    edge_scatter_exp<<<cdiv(E, 8), 256, 0, stream>>>(src, dst, et, alp, amax, xr,
                                                     denom, acc, E, N);
    bias_relu<<<cdiv((size_t)N * 128, 256), 256, 0, stream>>>(acc, denom, b, h, hb,
                                                              (size_t)N * 128);
  };

  cvt_f32_bf16<<<cdiv((size_t)N * 128, 256), 256, 0, stream>>>(x, xb, (size_t)N * 128);
  run_layer(xb, w0, q0, k0, b0);   // layer 0: output h (fp32) + hb (bf16)
  run_layer(hb, w1, q1, k1, b1);   // layer 1: reads hb in GEMM, overwrites it at the end

  fill_f32<<<1, 256, 0, stream>>>(psum, 0.f, 128);
  fill_f32<<<1, 256, 0, stream>>>(pmax, -INFINITY, 128);
  pool_kernel<<<512, 256, 0, stream>>>(h, psum, pmax, N);
  head_kernel<<<1, 256, 0, stream>>>(psum, pmax, fc1w, fc1b, fc2w, fc2b,
                                     (float*)d_out, N);
}
